// OuterProductMean_17875653886367
// MI455X (gfx1250) — matrix operand order repaired and checked by reading, not compiled, not run on hardware
//
#include <hip/hip_runtime.h>

// OuterProductMean: C[b] = A[b]^T B[b] / S,  A,B: (32, 4096, 256) fp32, C: (32, 256, 256) fp32
// bf16 hi/lo 3-term WMMA split (hi*hi + hi*lo + lo*hi), fp32 accumulate.
// 64x64 tile / workgroup, 4 waves (wave32), K chunked by 32,
// double-buffered LDS (1 barrier per chunk), b64 packed transpose stores,
// conflict-free stride-40 ds_load_b128 operand reads.

typedef __attribute__((ext_vector_type(16))) __bf16 v16bf;
typedef __attribute__((ext_vector_type(8)))  __bf16 v8bf;
typedef __attribute__((ext_vector_type(4)))  __bf16 v4bf;
typedef __attribute__((ext_vector_type(8)))  float  v8f;

#define S_DIM  4096
#define MN     256
#define KC     32                 // K per chunk (one bf16 WMMA depth)
#define TILE   64                 // output tile edge per workgroup
#define LDSTR  40                 // bf16 per LDS row (32 + 8 pad) = 80B
#define NCHUNK (S_DIM / KC)       // 128

__device__ __forceinline__ v16bf ld_op(const __bf16* p, int off0, int off1) {
    v8bf a = *(const v8bf*)(p + off0);
    v8bf b = *(const v8bf*)(p + off1);
    return __builtin_shufflevector(a, b, 0,1,2,3,4,5,6,7,8,9,10,11,12,13,14,15);
}

__global__ __launch_bounds__(128)
void OuterProductMean_17875653886367_kernel(const float* __restrict__ A,
                                            const float* __restrict__ B,
                                            float* __restrict__ C) {
    // double-buffered staging: 4 arrays x 2 bufs x 64x40 bf16 = 40 KB
    __shared__ __bf16 sAhi[2][TILE * LDSTR];
    __shared__ __bf16 sAlo[2][TILE * LDSTR];
    __shared__ __bf16 sBhi[2][TILE * LDSTR];
    __shared__ __bf16 sBlo[2][TILE * LDSTR];

    const int tid  = threadIdx.x;
    const int lane = tid & 31;
    const int wave = tid >> 5;

    const int tiles_per_b = (MN / TILE) * (MN / TILE);      // 16
    const int b      = blockIdx.x / tiles_per_b;
    const int tileid = blockIdx.x % tiles_per_b;
    const int tm = (tileid / (MN / TILE)) * TILE;
    const int tn = (tileid % (MN / TILE)) * TILE;

    const size_t boff = (size_t)b * S_DIM * MN;

    // staging mapping: quad = 4 consecutive m-columns, rp = 4 consecutive k-rows
    const int quad = tid & 15;
    const int rp   = tid >> 4;                               // 0..7 -> rows 4rp..4rp+3

    const float* Abase = A + boff + tm + 4 * quad;
    const float* Bbase = B + boff + tn + 4 * quad;

    // 2x2 wave grid over the 64x64 tile; each wave owns 2x2 16x16 subtiles
    const int wm = (wave >> 1) * 32;
    const int wn = (wave & 1)  * 32;

    // A operand (16-bit 16x32): lane<16 -> K 0..15 contiguous; lane>=16 -> K 16..31
    const int a_k0 = (lane < 16) ? 0 : 16;
    // B operand (16-bit 32x16): lane<16 -> K 0..15 contiguous; lane>=16 -> K 16..31
    const int b_k0 = (lane < 16) ? 0 : 16;
    const int lrow = lane & 15;

    v8f acc[2][2] = {};
    float4 ra[4], rb[4];

    // split fp32 -> (hi,lo) bf16 and store transposed [m][k] as packed b64 (4 k per store)
    auto stage = [&](int buf) {
        float av[16], bv[16];
        #pragma unroll
        for (int r = 0; r < 4; ++r) {
            av[4*r+0] = ra[r].x; av[4*r+1] = ra[r].y; av[4*r+2] = ra[r].z; av[4*r+3] = ra[r].w;
            bv[4*r+0] = rb[r].x; bv[4*r+1] = rb[r].y; bv[4*r+2] = rb[r].z; bv[4*r+3] = rb[r].w;
        }
        #pragma unroll
        for (int j = 0; j < 4; ++j) {
            v4bf ah, al, bh, bl;
            #pragma unroll
            for (int r = 0; r < 4; ++r) {
                float x  = av[4*r+j];
                __bf16 h = (__bf16)x;                 // RNE fp32 -> bf16
                ah[r] = h; al[r] = (__bf16)(x - (float)h);
                x  = bv[4*r+j];
                h  = (__bf16)x;
                bh[r] = h; bl[r] = (__bf16)(x - (float)h);
            }
            const int off = (4 * quad + j) * LDSTR + 4 * rp;  // 8B-aligned
            *(v4bf*)&sAhi[buf][off] = ah;
            *(v4bf*)&sAlo[buf][off] = al;
            *(v4bf*)&sBhi[buf][off] = bh;
            *(v4bf*)&sBlo[buf][off] = bl;
        }
    };

    // prologue: chunk 0 -> regs -> LDS buf 0
    #pragma unroll
    for (int r = 0; r < 4; ++r) {
        ra[r] = *(const float4*)(Abase + (size_t)(4 * rp + r) * MN);
        rb[r] = *(const float4*)(Bbase + (size_t)(4 * rp + r) * MN);
    }
    stage(0);

    for (int kc = 0; kc < NCHUNK; ++kc) {
        __syncthreads();                         // buf[cur] ready for everyone
        const int cur = kc & 1;

        // issue next chunk's global loads early (latency hidden by WMMAs)
        if (kc + 1 < NCHUNK) {
            const size_t so = (size_t)(kc + 1) * KC;
            #pragma unroll
            for (int r = 0; r < 4; ++r) {
                ra[r] = *(const float4*)(Abase + (so + 4 * rp + r) * MN);
                rb[r] = *(const float4*)(Bbase + (so + 4 * rp + r) * MN);
            }
        }

        // operand fetch from buf[cur] (conflict-free ds_load_b128 pairs)
        v16bf ahi[2], alo[2], bhi[2], blo[2];
        #pragma unroll
        for (int i = 0; i < 2; ++i) {
            const int mr = (wm + 16 * i + lrow) * LDSTR;
            ahi[i] = ld_op(&sAhi[cur][0], mr + a_k0, mr + a_k0 + 8);
            alo[i] = ld_op(&sAlo[cur][0], mr + a_k0, mr + a_k0 + 8);
            const int nr = (wn + 16 * i + lrow) * LDSTR;
            bhi[i] = ld_op(&sBhi[cur][0], nr + b_k0, nr + b_k0 + 8);
            blo[i] = ld_op(&sBlo[cur][0], nr + b_k0, nr + b_k0 + 8);
        }

        #pragma unroll
        for (int i = 0; i < 2; ++i)
            #pragma unroll
            for (int j = 0; j < 2; ++j) {
                acc[i][j] = __builtin_amdgcn_wmma_f32_16x16x32_bf16(
                    false, ahi[i], false, bhi[j], (short)0, acc[i][j], false, false);
                acc[i][j] = __builtin_amdgcn_wmma_f32_16x16x32_bf16(
                    false, ahi[i], false, blo[j], (short)0, acc[i][j], false, false);
                acc[i][j] = __builtin_amdgcn_wmma_f32_16x16x32_bf16(
                    false, alo[i], false, bhi[j], (short)0, acc[i][j], false, false);
            }

        // convert + store next chunk into the other buffer (co-executes with WMMAs)
        if (kc + 1 < NCHUNK) stage(cur ^ 1);
    }

    // epilogue: C/D layout — VGPR r: lanes 0-15 -> M=r, lanes 16-31 -> M=r+8; N = lane&15
    const float inv_s  = 1.0f / (float)S_DIM;
    const int hi_half  = (lane >> 4) * 8;
    #pragma unroll
    for (int i = 0; i < 2; ++i)
        #pragma unroll
        for (int j = 0; j < 2; ++j)
            #pragma unroll
            for (int r = 0; r < 8; ++r) {
                const int m = tm + wm + 16 * i + r + hi_half;
                const int n = tn + wn + 16 * j + (lane & 15);
                C[(size_t)b * MN * MN + (size_t)m * MN + n] = acc[i][j][r] * inv_s;
            }
}

extern "C" void kernel_launch(void* const* d_in, const int* in_sizes, int n_in,
                              void* d_out, int out_size, void* d_ws, size_t ws_size,
                              hipStream_t stream) {
    const float* A  = (const float*)d_in[0];
    const float* B  = (const float*)d_in[1];
    float*       Cp = (float*)d_out;
    dim3 grid(32 * (MN / TILE) * (MN / TILE));   // 512 workgroups
    dim3 block(128);                              // 4 waves (wave32)
    hipLaunchKernelGGL(OuterProductMean_17875653886367_kernel, grid, block, 0, stream,
                       A, B, Cp);
}
